// TempModule_74620761801330
// MI455X (gfx1250) — matrix operand order repaired and checked by reading, not compiled, not run on hardware
//
#include <hip/hip_runtime.h>
#include <hip/hip_bf16.h>

// ---------------------------------------------------------------------------
// Mamba block for MI455X (gfx1250, wave32, WMMA).
// All GEMMs use v_wmma_f32_16x16x32_bf16 (bf16 operands, f32 accumulate).
// Pipeline is HBM-bandwidth bound (~hundreds of MB @ 23.3 TB/s); scan is
// latency bound but 2048 wave32s of independent state hide it.
// B fragments are batch-loaded into distinct registers so WMMAs drain the
// load clause with partial s_wait_loadcnt (load/compute overlap).
// ---------------------------------------------------------------------------

#define DM     128   // D_MODEL
#define DI     256   // D_INNER
#define DS     16    // D_STATE
#define DTRK   8     // DT_RANK
#define DCONV  4
#define NGRP   4
#define BATCH  2
#define TLEN   256
#define NNODES 128
#define NSEQ   (BATCH * NNODES)   // 256
#define NTOK   (NSEQ * TLEN)      // 65536
#define GN_EPS 1e-5f

typedef __attribute__((ext_vector_type(16))) __bf16 v16bf;
typedef __attribute__((ext_vector_type(8)))  __bf16 v8bf;
typedef __attribute__((ext_vector_type(8)))  float  v8f;

// ---------------- workspace layout (bytes, all 256-aligned) ----------------
static constexpr size_t OFF_VBF   = 0;                                     // v in bf16
static constexpr size_t SZ_VBF    = (size_t)BATCH * TLEN * NNODES * DM * 2;
static constexpr size_t OFF_WIN   = OFF_VBF + SZ_VBF;                      // in_proj_w bf16 (512x128)
static constexpr size_t SZ_WIN    = (size_t)2 * DI * DM * 2;
static constexpr size_t OFF_WOUT  = OFF_WIN + SZ_WIN;                      // out_proj_w bf16 (128x256)
static constexpr size_t SZ_WOUT   = (size_t)DM * DI * 2;
static constexpr size_t OFF_WXP   = OFF_WOUT + SZ_WOUT;                    // x_proj_w bf16 padded (48x256)
static constexpr size_t SZ_WXP    = (size_t)48 * DI * 2;
static constexpr size_t OFF_XIN   = OFF_WXP + SZ_WXP;                      // x_in f32 (NTOK x DI); later reused as dt
static constexpr size_t SZ_TOKDI4 = (size_t)NTOK * DI * 4;
static constexpr size_t OFF_Z     = OFF_XIN + SZ_TOKDI4;                   // z f32
static constexpr size_t OFF_XCONV = OFF_Z + SZ_TOKDI4;                     // x_conv f32; later reused as prenorm out
static constexpr size_t OFF_XCBF  = OFF_XCONV + SZ_TOKDI4;                 // x_conv bf16; later reused as gated-y bf16
static constexpr size_t SZ_TOKDI2 = (size_t)NTOK * DI * 2;
static constexpr size_t OFF_DTR   = OFF_XCBF + SZ_TOKDI2;                  // dt_r f32 (NTOK x 8)
static constexpr size_t OFF_B     = OFF_DTR + (size_t)NTOK * DTRK * 4;     // B f32 (NTOK x 16)
static constexpr size_t OFF_C     = OFF_B + (size_t)NTOK * DS * 4;         // C f32 (NTOK x 16)
static constexpr size_t OFF_STATS = OFF_C + (size_t)NTOK * DS * 4;         // 8 groups x {sum, sumsq}
static constexpr size_t OFF_DUMP  = OFF_STATS + 256;                       // write-only sink for padded cols

// ---------------- WMMA fragment helpers (gfx1250 16x16x32 bf16) ------------
// A (16xK tile, row-major): lane half selects K-subrange; dwords 0-3 hold
// K = kb + half*8 + [0..7], dwords 4-7 hold K = kb + 16 + half*8 + [0..7].
__device__ __forceinline__ v16bf frag_a_load(const __bf16* rowptr, int kb, int half) {
  const __bf16* p0 = rowptr + kb + half * 8;
  v8bf lo = *reinterpret_cast<const v8bf*>(p0);
  v8bf hi = *reinterpret_cast<const v8bf*>(p0 + 16);
  v16bf r;
#pragma unroll
  for (int i = 0; i < 8; ++i) { r[i] = lo[i]; r[8 + i] = hi[i]; }
  return r;
}

// B (Kx16 tile) built from weight W[n][k] (row-major, leading dim ldk):
// lane = khalf*16 + n; lane holds W[n0+n][kb + khalf*8 + 0..7] and [kb + 16 + khalf*8 + 0..7].
__device__ __forceinline__ v16bf frag_b_load(const __bf16* W, int ldk, int n0, int kb, int lane) {
  int khalf = lane >> 4, n = lane & 15;
  const __bf16* p = W + (size_t)(n0 + n) * ldk + kb + khalf * 8;
  v8bf lo = *reinterpret_cast<const v8bf*>(p);
  v8bf hi = *reinterpret_cast<const v8bf*>(p + 16);
  v16bf r;
#pragma unroll
  for (int i = 0; i < 8; ++i) { r[i] = lo[i]; r[8 + i] = hi[i]; }
  return r;
}

// ---------------- small converters ----------------
__global__ void k_f32_to_bf16(const float* __restrict__ src, __bf16* __restrict__ dst, int n) {
  int i = blockIdx.x * blockDim.x + threadIdx.x;
  if (i < n) dst[i] = (__bf16)src[i];
}

__global__ void k_xprojw_pad(const float* __restrict__ src, __bf16* __restrict__ dst) {
  int i = blockIdx.x * blockDim.x + threadIdx.x;   // 48*256 elements
  if (i >= 48 * DI) return;
  int row = i / DI, col = i - row * DI;
  dst[i] = (__bf16)(row < (DTRK + 2 * DS) ? src[row * DI + col] : 0.f);
}

// ---------------- in_proj GEMM: (NTOK x 128) x (512 x 128)^T ---------------
// 1 block = 1 M-tile of 16 tokens; 4 waves, wave w covers N = w*128..+127.
// Token index = seq*TLEN + t (Mamba layout); source v is (b,t,n,d).
__global__ __launch_bounds__(128) void k_inproj(const __bf16* __restrict__ vbf,
                                                const __bf16* __restrict__ wbf,
                                                float* __restrict__ x_in,
                                                float* __restrict__ z) {
  int lane = threadIdx.x & 31, wave = threadIdx.x >> 5;
  int half = lane >> 4;
  int row0 = blockIdx.x * 16;
  int m = row0 + (lane & 15);
  int s = m >> 8, t = m & 255;       // tok = s*TLEN + t
  int b = s >> 7, nn = s & 127;
  const __bf16* arow = vbf + (((size_t)(b * TLEN + t) * NNODES + nn) * DM);

  v16bf a[4];
#pragma unroll
  for (int kk = 0; kk < 4; ++kk) a[kk] = frag_a_load(arow, kk * 32, half);

  int n_base = wave * 128;
  size_t rbase = (size_t)(row0 + half * 8) * DI;   // fragment row base (x DI)
#pragma unroll
  for (int ns = 0; ns < 8; ++ns) {
    int n0 = n_base + ns * 16;
    if (ns < 7)  // warm GL2 for the next weight sub-tile
      __builtin_prefetch(wbf + (size_t)(n0 + 16) * DM, 0, 1);
    // batch-load all 4 B fragments so the WMMA chain drains them with
    // partial s_wait_loadcnt (overlap loads with matrix ops)
    v16bf bfr[4];
#pragma unroll
    for (int kk = 0; kk < 4; ++kk)
      bfr[kk] = frag_b_load(wbf, DM, n0, kk * 32, lane);
    v8f acc = {};
#pragma unroll
    for (int kk = 0; kk < 4; ++kk)
      acc = __builtin_amdgcn_wmma_f32_16x16x32_bf16(false, a[kk], false, bfr[kk],
                                                    (short)0, acc, false, false);
    // branchless split store: select destination pointer per lane
    int col = n0 + (lane & 15);
    float* dst = (col < DI) ? (x_in + rbase + col) : (z + rbase + (col - DI));
#pragma unroll
    for (int r = 0; r < 8; ++r)
      dst[(size_t)r * DI] = acc[r];
  }
}

// ---------------- depthwise conv (K=4) + SiLU ------------------------------
__global__ void k_conv_silu(const float* __restrict__ x_in,
                            const float* __restrict__ conv_w,
                            const float* __restrict__ conv_b,
                            float* __restrict__ xconv,
                            __bf16* __restrict__ xconv_bf) {
  int idx = blockIdx.x * blockDim.x + threadIdx.x;   // NTOK*DI
  int e = idx & (DI - 1);
  int tok = idx >> 8;
  int s = tok >> 8, t = tok & 255;
  float acc = conv_b[e];
#pragma unroll
  for (int k = 0; k < DCONV; ++k) {
    int tk = t - (DCONV - 1) + k;
    if (tk >= 0)
      acc = fmaf(x_in[((size_t)s * TLEN + tk) * DI + e], conv_w[e * DCONV + k], acc);
  }
  float y = acc / (1.f + __expf(-acc));   // silu
  xconv[idx] = y;
  xconv_bf[idx] = (__bf16)y;
}

// ---------------- x_proj GEMM: (NTOK x 256) x (48 x 256)^T -----------------
// 4 waves/block, each wave owns one M-tile; 3 N-subtiles, branchless scatter.
__global__ __launch_bounds__(128) void k_xproj(const __bf16* __restrict__ xcbf,
                                               const __bf16* __restrict__ wxp,
                                               float* __restrict__ dtr,
                                               float* __restrict__ Bb,
                                               float* __restrict__ Cb,
                                               float* __restrict__ dump) {
  int lane = threadIdx.x & 31, wave = threadIdx.x >> 5;
  int half = lane >> 4;
  int mt = blockIdx.x * 4 + wave;
  int row0 = mt * 16;
  const __bf16* arow = xcbf + (size_t)(row0 + (lane & 15)) * DI;

  v16bf a[8];
#pragma unroll
  for (int kk = 0; kk < 8; ++kk) a[kk] = frag_a_load(arow, kk * 32, half);

  int rbase = row0 + half * 8;
#pragma unroll
  for (int ns = 0; ns < 3; ++ns) {
    int n0 = ns * 16;
    v8f acc = {};
#pragma unroll
    for (int kc = 0; kc < 2; ++kc) {        // two chunks of 4 fragments
      v16bf bfr[4];
#pragma unroll
      for (int kk = 0; kk < 4; ++kk)
        bfr[kk] = frag_b_load(wxp, DI, n0, (kc * 4 + kk) * 32, lane);
#pragma unroll
      for (int kk = 0; kk < 4; ++kk)
        acc = __builtin_amdgcn_wmma_f32_16x16x32_bf16(false, a[kc * 4 + kk], false,
                                                      bfr[kk], (short)0, acc,
                                                      false, false);
    }
    // branchless 4-way scatter: select pointer + row stride per lane.
    // Padded cols (40..47) write to a never-read sink with stride 0.
    int col = n0 + (lane & 15);
    bool isdt = col < DTRK;
    bool isB  = col < DTRK + DS;
    bool isC  = col < DTRK + 2 * DS;
    float* dst = isdt ? (dtr + (size_t)rbase * DTRK + col)
               : isB  ? (Bb + (size_t)rbase * DS + (col - DTRK))
               : isC  ? (Cb + (size_t)rbase * DS + (col - DTRK - DS))
                      : (dump + (col - DTRK - 2 * DS));
    size_t ld = isdt ? DTRK : (isC ? DS : (isB ? DS : 0));
#pragma unroll
    for (int r = 0; r < 8; ++r)
      dst[(size_t)r * ld] = acc[r];
  }
}

// ---------------- dt = softplus(dt_r @ dtW^T + b) --------------------------
__global__ void k_dt(const float* __restrict__ dtr,
                     const float* __restrict__ dtw,
                     const float* __restrict__ dtb,
                     float* __restrict__ dt) {
  int idx = blockIdx.x * blockDim.x + threadIdx.x;   // NTOK*DI
  int e = idx & (DI - 1);
  int tok = idx >> 8;
  float acc = dtb[e];
#pragma unroll
  for (int r = 0; r < DTRK; ++r)
    acc = fmaf(dtr[(size_t)tok * DTRK + r], dtw[e * DTRK + r], acc);
  dt[idx] = (acc > 20.f) ? acc : log1pf(__expf(acc));
}

// ---------------- selective scan + D-skip + SiLU(z) gating -----------------
// 1 block = 1 sequence; thread = channel e; h[16] in VGPRs.
// B/C broadcast via double-buffered LDS: one barrier per timestep.
__global__ __launch_bounds__(DI) void k_scan(const float* __restrict__ dt,
                                             const float* __restrict__ xconv,
                                             const float* __restrict__ z,
                                             const float* __restrict__ Bb,
                                             const float* __restrict__ Cb,
                                             const float* __restrict__ A_log,
                                             const float* __restrict__ D_param,
                                             __bf16* __restrict__ ybf) {
  __shared__ float lB[2][DS], lC[2][DS];
  int s = blockIdx.x;
  int e = threadIdx.x;

  float a[DS];
#pragma unroll
  for (int j = 0; j < DS; ++j) a[j] = -__expf(A_log[e * DS + j]);
  float h[DS];
#pragma unroll
  for (int j = 0; j < DS; ++j) h[j] = 0.f;
  float Dp = D_param[e];

  size_t tok0 = (size_t)s * TLEN;
  // stage t = 0
  if (e < DS)            lB[0][e] = Bb[tok0 * DS + e];
  else if (e < 2 * DS)   lC[0][e - DS] = Cb[tok0 * DS + (e - DS)];
  __syncthreads();

  for (int t = 0; t < TLEN; ++t) {
    int buf = t & 1;
    // stage t+1 into the other buffer (consumed after the barrier below)
    if (t + 1 < TLEN) {
      size_t tok1 = tok0 + t + 1;
      if (e < DS)          lB[buf ^ 1][e] = Bb[tok1 * DS + e];
      else if (e < 2 * DS) lC[buf ^ 1][e - DS] = Cb[tok1 * DS + (e - DS)];
    }

    size_t tok = tok0 + t;
    float dtv = dt[tok * DI + e];
    float xv  = xconv[tok * DI + e];
    float zv  = z[tok * DI + e];
    float dx = dtv * xv;
    float y = 0.f;
#pragma unroll
    for (int j = 0; j < DS; ++j) {
      float dA = __expf(dtv * a[j]);
      h[j] = fmaf(dA, h[j], dx * lB[buf][j]);
      y = fmaf(h[j], lC[buf][j], y);
    }
    y = fmaf(Dp, xv, y);
    float gate = zv / (1.f + __expf(-zv));
    ybf[tok * DI + e] = (__bf16)(y * gate);
    __syncthreads();
  }
}

// ---------------- out_proj GEMM + residual ---------------------------------
__global__ __launch_bounds__(128) void k_outproj(const __bf16* __restrict__ ybf,
                                                 const __bf16* __restrict__ wout,
                                                 const float* __restrict__ v,
                                                 float* __restrict__ pre) {
  int lane = threadIdx.x & 31, wave = threadIdx.x >> 5;
  int half = lane >> 4;
  int mt = blockIdx.x * 4 + wave;
  int row0 = mt * 16;
  const __bf16* arow = ybf + (size_t)(row0 + (lane & 15)) * DI;

  v16bf a[8];
#pragma unroll
  for (int kk = 0; kk < 8; ++kk) a[kk] = frag_a_load(arow, kk * 32, half);

  // within a fragment the 8 rows share (s,b,nn); only t advances by r
  int rbase = row0 + half * 8;              // tok = s*TLEN + t
  int s = rbase >> 8, tb = rbase & 255;
  int b = s >> 7, nn = s & 127;
  size_t gbase = ((size_t)(b * TLEN + tb) * NNODES + nn) * DM;

#pragma unroll
  for (int ns = 0; ns < 8; ++ns) {
    int n0 = ns * 16;
    v8f acc = {};
#pragma unroll
    for (int kc = 0; kc < 2; ++kc) {        // two chunks of 4 fragments
      v16bf bfr[4];
#pragma unroll
      for (int kk = 0; kk < 4; ++kk)
        bfr[kk] = frag_b_load(wout, DI, n0, (kc * 4 + kk) * 32, lane);
#pragma unroll
      for (int kk = 0; kk < 4; ++kk)
        acc = __builtin_amdgcn_wmma_f32_16x16x32_bf16(false, a[kc * 4 + kk], false,
                                                      bfr[kk], (short)0, acc,
                                                      false, false);
    }
    int d = n0 + (lane & 15);
    size_t gi0 = gbase + d;
#pragma unroll
    for (int r = 0; r < 8; ++r) {
      size_t gi = gi0 + (size_t)r * (NNODES * DM);
      pre[gi] = acc[r] + v[gi];
    }
  }
}

// ---------------- group norm -----------------------------------------------
__global__ void k_gn_zero(float* __restrict__ stats) {
  if (threadIdx.x < 2 * BATCH * NGRP) stats[threadIdx.x] = 0.f;
}

__global__ __launch_bounds__(256) void k_gn_reduce(const float* __restrict__ pre,
                                                   float* __restrict__ stats) {
  __shared__ float ssum[256], ssq[256];
  int bg  = blockIdx.x >> 9;          // 8 (batch,group) pairs x 512 blocks
  int blk = blockIdx.x & 511;
  int b = bg >> 2, g = bg & 3;
  int tid = threadIdx.x;
  float s1 = 0.f, s2 = 0.f;
#pragma unroll
  for (int i = 0; i < 8; ++i) {
    int li = blk * 2048 + i * 256 + tid;          // over t*n*c = 1,048,576
    int c = li & 31, n = (li >> 5) & 127, t = li >> 12;
    float x = pre[((size_t)(b * TLEN + t) * NNODES + n) * DM + g * 32 + c];
    s1 += x;
    s2 = fmaf(x, x, s2);
  }
  ssum[tid] = s1; ssq[tid] = s2;
  __syncthreads();
  for (int off = 128; off > 0; off >>= 1) {
    if (tid < off) { ssum[tid] += ssum[tid + off]; ssq[tid] += ssq[tid + off]; }
    __syncthreads();
  }
  if (tid == 0) {
    atomicAdd(&stats[bg * 2 + 0], ssum[0]);
    atomicAdd(&stats[bg * 2 + 1], ssq[0]);
  }
}

__global__ void k_gn_norm(const float* __restrict__ pre,
                          const float* __restrict__ stats,
                          const float* __restrict__ gamma,
                          const float* __restrict__ beta,
                          float* __restrict__ out) {
  int idx = blockIdx.x * blockDim.x + threadIdx.x;   // b*t*n*f
  int d = idx & 127;
  int b = idx >> 22;
  int g = d >> 5;
  int bg = b * NGRP + g;
  const float invN = 1.f / (float)(TLEN * NNODES * (DM / NGRP));
  float mean = stats[bg * 2 + 0] * invN;
  float var  = stats[bg * 2 + 1] * invN - mean * mean;
  float x = pre[idx];
  out[idx] = (x - mean) * rsqrtf(var + GN_EPS) * gamma[d] + beta[d];
}

// ---------------------------------------------------------------------------
extern "C" void kernel_launch(void* const* d_in, const int* in_sizes, int n_in,
                              void* d_out, int out_size, void* d_ws, size_t ws_size,
                              hipStream_t stream) {
  (void)in_sizes; (void)n_in; (void)out_size; (void)ws_size;
  const float* v         = (const float*)d_in[0];
  const float* in_proj_w = (const float*)d_in[1];
  const float* conv_w    = (const float*)d_in[2];
  const float* conv_b    = (const float*)d_in[3];
  const float* x_proj_w  = (const float*)d_in[4];
  const float* dt_proj_w = (const float*)d_in[5];
  const float* dt_proj_b = (const float*)d_in[6];
  const float* A_log     = (const float*)d_in[7];
  const float* D_param   = (const float*)d_in[8];
  const float* out_proj_w= (const float*)d_in[9];
  const float* gn_gamma  = (const float*)d_in[10];
  const float* gn_beta   = (const float*)d_in[11];
  float* out = (float*)d_out;

  char* ws = (char*)d_ws;
  __bf16* vbf   = (__bf16*)(ws + OFF_VBF);
  __bf16* winbf = (__bf16*)(ws + OFF_WIN);
  __bf16* woutbf= (__bf16*)(ws + OFF_WOUT);
  __bf16* wxpbf = (__bf16*)(ws + OFF_WXP);
  float*  x_in  = (float*)(ws + OFF_XIN);
  float*  dt    = (float*)(ws + OFF_XIN);     // alias: x_in dead after conv
  float*  z     = (float*)(ws + OFF_Z);
  float*  xconv = (float*)(ws + OFF_XCONV);
  float*  pre   = (float*)(ws + OFF_XCONV);   // alias: xconv dead after scan
  __bf16* xcbf  = (__bf16*)(ws + OFF_XCBF);
  __bf16* ybf   = (__bf16*)(ws + OFF_XCBF);   // alias: xconv_bf dead after x_proj
  float*  dtr   = (float*)(ws + OFF_DTR);
  float*  Bb    = (float*)(ws + OFF_B);
  float*  Cb    = (float*)(ws + OFF_C);
  float*  stats = (float*)(ws + OFF_STATS);
  float*  dump  = (float*)(ws + OFF_DUMP);

  const int nv = BATCH * TLEN * NNODES * DM;           // 8,388,608

  // 1) precision converts
  k_f32_to_bf16<<<(nv + 255) / 256, 256, 0, stream>>>(v, vbf, nv);
  k_f32_to_bf16<<<(2 * DI * DM + 255) / 256, 256, 0, stream>>>(in_proj_w, winbf, 2 * DI * DM);
  k_f32_to_bf16<<<(DM * DI + 255) / 256, 256, 0, stream>>>(out_proj_w, woutbf, DM * DI);
  k_xprojw_pad<<<(48 * DI + 255) / 256, 256, 0, stream>>>(x_proj_w, wxpbf);

  // 2) in_proj GEMM -> x_in, z   (M=65536 tokens, N=512, K=128)
  k_inproj<<<NTOK / 16, 128, 0, stream>>>(vbf, winbf, x_in, z);

  // 3) depthwise conv + SiLU
  k_conv_silu<<<(NTOK * DI) / 256, 256, 0, stream>>>(x_in, conv_w, conv_b, xconv, xcbf);

  // 4) x_proj GEMM -> dt_r, B, C (N padded 40->48)
  k_xproj<<<NTOK / 64, 128, 0, stream>>>(xcbf, wxpbf, dtr, Bb, Cb, dump);

  // 5) dt projection + softplus (writes over x_in buffer)
  k_dt<<<(NTOK * DI) / 256, 256, 0, stream>>>(dtr, dt_proj_w, dt_proj_b, dt);

  // 6) selective scan, fused D-skip + gating; emits bf16 y
  k_scan<<<NSEQ, DI, 0, stream>>>(dt, xconv, z, Bb, Cb, A_log, D_param, ybf);

  // 7) out_proj GEMM + residual -> prenorm (v layout)
  k_outproj<<<NTOK / 64, 128, 0, stream>>>(ybf, woutbf, v, pre);

  // 8) group norm
  k_gn_zero<<<1, 32, 0, stream>>>(stats);
  k_gn_reduce<<<BATCH * NGRP * 512, 256, 0, stream>>>(pre, stats);
  k_gn_norm<<<nv / 256, 256, 0, stream>>>(pre, stats, gn_gamma, gn_beta, out);
}
